// UNetTransformer_32710470926929
// MI455X (gfx1250) — matrix operand order repaired and checked by reading, not compiled, not run on hardware
//
#include <hip/hip_runtime.h>
#include <hip/hip_bf16.h>
#include <math.h>
#include <stdint.h>

// ---------------------------------------------------------------------------
// CDNA5 (gfx1250) wave32 WMMA types
// ---------------------------------------------------------------------------
typedef _Float16 half_t;
typedef __attribute__((ext_vector_type(16))) _Float16 v16h;
typedef __attribute__((ext_vector_type(8)))  _Float16 v8h;
typedef __attribute__((ext_vector_type(8)))  float    v8f;
typedef __attribute__((ext_vector_type(4)))  float    v4f;

#define DEVFN static __device__ __forceinline__

DEVFN v8f wmma_f16(v16h a, v16h b, v8f c) {
  return __builtin_amdgcn_wmma_f32_16x16x32_f16(
      false, a, false, b, (short)0, c, false, false);
}

// A-matrix (16x32 f16), ISA 7.12.2. Caller passes p = row_base + 8*hf (f32 src).
DEVFN v16h make_a(const float* p) {
  v4f u0 = *(const v4f*)(p);
  v4f u1 = *(const v4f*)(p + 4);
  v4f u2 = *(const v4f*)(p + 16);
  v4f u3 = *(const v4f*)(p + 20);
  v16h a;
#pragma unroll
  for (int e = 0; e < 4; ++e) { a[e] = (half_t)u0[e]; a[4+e] = (half_t)u1[e];
                                a[8+e] = (half_t)u2[e]; a[12+e] = (half_t)u3[e]; }
  return a;
}
// B-matrix (32x16 f16). Caller passes p = row_base + 8*hf (f32 src).
DEVFN v16h make_b(const float* p) {
  v4f u0 = *(const v4f*)(p);
  v4f u1 = *(const v4f*)(p + 4);
  v4f u2 = *(const v4f*)(p + 16);
  v4f u3 = *(const v4f*)(p + 20);
  v16h b;
#pragma unroll
  for (int e = 0; e < 4; ++e) { b[e] = (half_t)u0[e]; b[4+e] = (half_t)u1[e];
                                b[8+e] = (half_t)u2[e]; b[12+e] = (half_t)u3[e]; }
  return b;
}
// Same layouts from packed-f16 sources (no conversion).
DEVFN v16h make_a_h(const half_t* p) {
  v8h lo = *(const v8h*)(p);        // K = 8*hf + 0..7
  v8h hi = *(const v8h*)(p + 16);   // K = 16 + 8*hf + 0..7
  return __builtin_shufflevector(lo, hi, 0,1,2,3,4,5,6,7,8,9,10,11,12,13,14,15);
}
DEVFN v16h make_b_h(const half_t* p) {
  v8h lo = *(const v8h*)(p);
  v8h hi = *(const v8h*)(p + 16);
  return __builtin_shufflevector(lo, hi, 0,1,2,3,4,5,6,7,8,9,10,11,12,13,14,15);
}

// async copy: 16 bytes per lane, global -> LDS (tracked by ASYNCcnt)
DEVFN void async_copy_b128(unsigned lds_addr, const void* gaddr) {
  asm volatile("global_load_async_to_lds_b128 %0, %1, off"
               :: "v"(lds_addr), "v"((unsigned long long)(uintptr_t)gaddr)
               : "memory");
}
template <int N> DEVFN void wait_async_le() {
  asm volatile("s_wait_asynccnt %0" :: "i"(N) : "memory");
}

// xor-lane exchange via ds_swizzle_b32 (imm pattern, no VALU address setup)
template <int XM> DEVFN float swz_xor_f(float x) {
  return __int_as_float(__builtin_amdgcn_ds_swizzle(__float_as_int(x), (XM << 10) | 0x1F));
}
template <int XM> DEVFN int swz_xor_i(int x) {
  return __builtin_amdgcn_ds_swizzle(x, (XM << 10) | 0x1F);
}
DEVFN float grp16_max(float x) {  // reduce across 16-lane group
  x = fmaxf(x, swz_xor_f<8>(x)); x = fmaxf(x, swz_xor_f<4>(x));
  x = fmaxf(x, swz_xor_f<2>(x)); x = fmaxf(x, swz_xor_f<1>(x));
  return x;
}
DEVFN float grp16_sum(float x) {
  x += swz_xor_f<8>(x); x += swz_xor_f<4>(x);
  x += swz_xor_f<2>(x); x += swz_xor_f<1>(x);
  return x;
}
DEVFN float wave_sum(float x) {
  x += swz_xor_f<16>(x); x += swz_xor_f<8>(x); x += swz_xor_f<4>(x);
  x += swz_xor_f<2>(x); x += swz_xor_f<1>(x);
  return x;
}
DEVFN float fast_exp(float x) {           // v_exp_f32
  return __builtin_amdgcn_exp2f(x * 1.4426950408889634f);
}

// ---------------------------------------------------------------------------
// GEMM (NT): C[b][m][n] = sum_k A[b][m][k] * W[n][k] + bias[n] (+resid)
// Shapes compile-time => fully unrolled K loop, immediate-offset addressing.
// ---------------------------------------------------------------------------
template <int K, int Nc, bool OutHalf>
__global__ __launch_bounds__(32)
void k_gemm_nt(const float* __restrict__ A, const float* __restrict__ W,
               const float* __restrict__ bias, const float* __restrict__ resid,
               void* __restrict__ Cout, int M, int storeT) {
  const int lane = threadIdx.x & 31;
  const int lane16 = lane & 15, hf = lane >> 4;
  const int m0 = blockIdx.x * 16;
  const int n0 = blockIdx.y * 16;
  const float* Ab = A + (size_t)blockIdx.z * M * K;

  v8f acc;
  const float bv = bias ? bias[n0 + lane16] : 0.f;
#pragma unroll
  for (int r = 0; r < 8; ++r) acc[r] = bv;

  const float* pa = Ab + (size_t)(m0 + lane16) * K + 8 * hf;
  const float* pw = W + (size_t)(n0 + lane16) * K + 8 * hf;
#pragma unroll
  for (int k0 = 0; k0 < K; k0 += 32) {
    v16h a = make_a(pa + k0);
    v16h b = make_b(pw + k0);
    acc = wmma_f16(a, b, acc);
  }

  if (resid) {
    const float* Rb = resid + (size_t)blockIdx.z * M * Nc +
                      (size_t)(m0 + 8 * hf) * Nc + n0 + lane16;
#pragma unroll
    for (int r = 0; r < 8; ++r) acc[r] += Rb[r * Nc];
  }
  if constexpr (OutHalf) {
    half_t* Ch = (half_t*)Cout + (size_t)blockIdx.z * M * Nc +
                 (size_t)(m0 + 8 * hf) * Nc + n0 + lane16;
#pragma unroll
    for (int r = 0; r < 8; ++r) Ch[r * Nc] = (half_t)acc[r];
  } else {
    float* Cf = (float*)Cout + (size_t)blockIdx.z * M * Nc;
    if (!storeT) {
      float* p = Cf + (size_t)(m0 + 8 * hf) * Nc + n0 + lane16;
#pragma unroll
      for (int r = 0; r < 8; ++r) p[r * Nc] = acc[r];
    } else {  // C[b][n][m] (final layer -> [B, COUT, N])
      float* p = Cf + (size_t)(n0 + lane16) * M + m0 + 8 * hf;
#pragma unroll
      for (int r = 0; r < 8; ++r) p[r] = acc[r];
    }
  }
}

// ---------------------------------------------------------------------------
// Fused flash attention, f16 Q/K/V in [B,Np,D], f32 out. 4 waves / 64 queries
// per block. Double-buffered 32-key K/V tiles staged with
// global_load_async_to_lds_b128; s_wait_asynccnt NOUT overlaps tile i+1 fetch
// with tile i compute (async loads complete in order).
// ---------------------------------------------------------------------------
template <int D>
__global__ __launch_bounds__(128)
void k_attn(const half_t* __restrict__ Q, const half_t* __restrict__ Km,
            const half_t* __restrict__ V, float* __restrict__ O, int Np) {
  constexpr int ND32 = D / 32, ND16 = D / 16;
  constexpr int TILE = 32 * D;            // halves per tensor tile
  constexpr int CH = (TILE * 2) / 16;     // 16B chunks per tensor tile (mult of 128)
  constexpr int NOUT = 2 * (CH / 128);    // per-thread async issues per tile pair

  extern __shared__ char smem[];
  half_t* shT = (half_t*)smem;            // [2 bufs][K,V][TILE]
  half_t* shPall = shT + 4 * TILE;        // [4][16*32]

  const int tid = threadIdx.x;
  const int lane = tid & 31, wv = tid >> 5;
  const int lane16 = lane & 15, hf = lane >> 4;
  half_t* shP = shPall + wv * (16 * 32);
  const int m0 = blockIdx.x * 64 + wv * 16;
  const size_t bs = (size_t)Np * D;
  const half_t* Qb = Q + bs * blockIdx.y;
  const half_t* Kb = Km + bs * blockIdx.y;
  const half_t* Vb = V + bs * blockIdx.y;
  float* Ob = O + bs * blockIdx.y;

  v16h qreg[ND32];
#pragma unroll
  for (int c = 0; c < ND32; ++c)
    qreg[c] = make_a_h(Qb + (size_t)(m0 + lane16) * D + c * 32 + 8 * hf);

  float mrun[8], lrun[8];
  v8f acc[ND16];
#pragma unroll
  for (int r = 0; r < 8; ++r) { mrun[r] = -3e38f; lrun[r] = 0.f; }
#pragma unroll
  for (int j = 0; j < ND16; ++j)
#pragma unroll
    for (int r = 0; r < 8; ++r) acc[j][r] = 0.f;

  const unsigned ldsT = (unsigned)(uintptr_t)shT;  // low 32 bits = LDS offset
  auto issue_tile = [&](int kt, int b) {
    const half_t* gK = Kb + (size_t)kt * D;
    const half_t* gV = Vb + (size_t)kt * D;
    const unsigned dK = ldsT + (unsigned)b * (2 * TILE * 2);
    const unsigned dV = dK + TILE * 2;
#pragma unroll 1
    for (int c = tid; c < CH; c += 128) {
      async_copy_b128(dK + 16u * (unsigned)c, (const char*)gK + 16 * (size_t)c);
      async_copy_b128(dV + 16u * (unsigned)c, (const char*)gV + 16 * (size_t)c);
    }
  };

  issue_tile(0, 0);

  for (int k0 = 0; k0 < Np; k0 += 32) {
    const int cur = (k0 >> 5) & 1;
    if (k0 + 32 < Np) { issue_tile(k0 + 32, cur ^ 1); wait_async_le<NOUT>(); }
    else              { wait_async_le<0>(); }
    __syncthreads();
    const half_t* sK = shT + cur * (2 * TILE);
    const half_t* sV = sK + TILE;

    // S = Q * K^T for 32 keys (two 16x16 C-layout tiles), K read from LDS
    v8f s0, s1;
#pragma unroll
    for (int r = 0; r < 8; ++r) { s0[r] = 0.f; s1[r] = 0.f; }
#pragma unroll
    for (int c = 0; c < ND32; ++c) {
      v16h b0 = make_b_h(sK + lane16 * D + c * 32 + 8 * hf);
      v16h b1 = make_b_h(sK + (16 + lane16) * D + c * 32 + 8 * hf);
      s0 = wmma_f16(qreg[c], b0, s0);
      s1 = wmma_f16(qreg[c], b1, s1);
    }

    // Online softmax; row M=r+8*hf lives across one 16-lane group.
    float corr[8];
#pragma unroll
    for (int r = 0; r < 8; ++r) {
      float rm = grp16_max(fmaxf(s0[r], s1[r]));
      float mn = fmaxf(mrun[r], rm);
      corr[r] = fast_exp(mrun[r] - mn);
      float e0 = fast_exp(s0[r] - mn), e1 = fast_exp(s1[r] - mn);
      s0[r] = e0; s1[r] = e1;
      float ps = grp16_sum(e0 + e1);
      lrun[r] = lrun[r] * corr[r] + ps;
      mrun[r] = mn;
    }

    // P (16x32) C-layout -> LDS -> A-layout (per-wave staging buffer)
#pragma unroll
    for (int r = 0; r < 8; ++r) {
      int row = r + 8 * hf;
      shP[row * 32 + lane16] = (half_t)s0[r];
      shP[row * 32 + 16 + lane16] = (half_t)s1[r];
    }
    __syncthreads();
    v16h pa = make_a_h(shP + lane16 * 32 + 8 * hf);

    // acc = acc*corr + P * V_tile ; V read from LDS with immediate offsets
#pragma unroll
    for (int j = 0; j < ND16; ++j) {
      const half_t* vp = sV + (8 * hf) * D + j * 16 + lane16;
      v16h bvv;
#pragma unroll
      for (int e = 0; e < 16; ++e) bvv[e] = vp[(e < 8 ? e : e + 8) * D];
      v8f t = acc[j];
#pragma unroll
      for (int r = 0; r < 8; ++r) t[r] *= corr[r];
      acc[j] = wmma_f16(pa, bvv, t);
    }
    __syncthreads();  // all reads of buf `cur` done before it is re-filled
  }

  float invl[8];
#pragma unroll
  for (int r = 0; r < 8; ++r) invl[r] = 1.0f / lrun[r];
#pragma unroll
  for (int j = 0; j < ND16; ++j)
#pragma unroll
    for (int r = 0; r < 8; ++r)
      Ob[(size_t)(m0 + r + 8 * hf) * D + j * 16 + lane16] = acc[j][r] * invl[r];
}

// ---------------------------------------------------------------------------
// Farthest point sampling: one workgroup per batch; xyz+dist resident in LDS.
// ---------------------------------------------------------------------------
DEVFN void amax_step(float& v, int& i, float v2, int i2) {
  if (v2 > v || (v2 == v && i2 < i)) { v = v2; i = i2; }
}
DEVFN int block_argmax(float v, int i, float* rv, int* ri, int nwaves) {
  const int lane = threadIdx.x & 31, w = threadIdx.x >> 5;
  amax_step(v, i, swz_xor_f<16>(v), swz_xor_i<16>(i));
  amax_step(v, i, swz_xor_f<8>(v),  swz_xor_i<8>(i));
  amax_step(v, i, swz_xor_f<4>(v),  swz_xor_i<4>(i));
  amax_step(v, i, swz_xor_f<2>(v),  swz_xor_i<2>(i));
  amax_step(v, i, swz_xor_f<1>(v),  swz_xor_i<1>(i));
  if (lane == 0) { rv[w] = v; ri[w] = i; }
  __syncthreads();
  if (w == 0) {
    float vv = (lane < nwaves) ? rv[lane] : -3.4e38f;
    int   ii = (lane < nwaves) ? ri[lane] : 0x7fffffff;
    amax_step(vv, ii, swz_xor_f<16>(vv), swz_xor_i<16>(ii));
    amax_step(vv, ii, swz_xor_f<8>(vv),  swz_xor_i<8>(ii));
    amax_step(vv, ii, swz_xor_f<4>(vv),  swz_xor_i<4>(ii));
    amax_step(vv, ii, swz_xor_f<2>(vv),  swz_xor_i<2>(ii));
    amax_step(vv, ii, swz_xor_f<1>(vv),  swz_xor_i<1>(ii));
    if (lane == 0) ri[0] = ii;
  }
  __syncthreads();
  int res = ri[0];
  __syncthreads();
  return res;
}

__global__ __launch_bounds__(1024)
void k_fps(const float* __restrict__ pts, int N, int npoint, int* __restrict__ idx) {
  extern __shared__ float sm[];
  float* sx = sm; float* sy = sm + N; float* sz = sm + 2 * N; float* dist = sm + 3 * N;
  __shared__ float rv[32]; __shared__ int ri[32]; __shared__ float csum[3];
  const float* P = pts + (size_t)blockIdx.x * N * 3;
  const int t = threadIdx.x, nt = blockDim.x;
  for (int i = t; i < N; i += nt) {
    sx[i] = P[3 * i]; sy[i] = P[3 * i + 1]; sz[i] = P[3 * i + 2]; dist[i] = 1e10f;
  }
  if (t < 3) csum[t] = 0.f;
  __syncthreads();
  float a0 = 0, a1 = 0, a2 = 0;
  for (int i = t; i < N; i += nt) { a0 += sx[i]; a1 += sy[i]; a2 += sz[i]; }
  a0 = wave_sum(a0); a1 = wave_sum(a1); a2 = wave_sum(a2);
  if ((t & 31) == 0) { atomicAdd(&csum[0], a0); atomicAdd(&csum[1], a1); atomicAdd(&csum[2], a2); }
  __syncthreads();
  const float cx = csum[0] / N, cy = csum[1] / N, cz = csum[2] / N;
  float bv = -3.4e38f; int bi = 0x7fffffff;
  for (int i = t; i < N; i += nt) {
    float dx = sx[i] - cx, dy = sy[i] - cy, dz = sz[i] - cz;
    float d = -(dx * dx + dy * dy + dz * dz);
    if (d > bv || (d == bv && i < bi)) { bv = d; bi = i; }
  }
  int far = block_argmax(bv, bi, rv, ri, nt >> 5);
  for (int it = 0; it < npoint; ++it) {
    if (t == 0) idx[(size_t)blockIdx.x * npoint + it] = far;
    const float fx = sx[far], fy = sy[far], fz = sz[far];
    float lv = -3.4e38f; int li = 0x7fffffff;
    for (int i = t; i < N; i += nt) {
      float dx = sx[i] - fx, dy = sy[i] - fy, dz = sz[i] - fz;
      float d = dx * dx + dy * dy + dz * dz;
      float nd = fminf(dist[i], d); dist[i] = nd;
      if (nd > lv || (nd == lv && i < li)) { lv = nd; li = i; }
    }
    far = block_argmax(lv, li, rv, ri, nt >> 5);
  }
}

// ---------------------------------------------------------------------------
// Elementwise helpers
// ---------------------------------------------------------------------------
__global__ void k_transpose_cn(const float* __restrict__ src, float* __restrict__ dst,
                               int C, int N) {  // [B,C,N] -> [B,N,C]
  int e = blockIdx.x * blockDim.x + threadIdx.x;
  if (e >= C * N) return;
  int c = e / N, n = e % N;
  dst[(size_t)blockIdx.y * N * C + (size_t)n * C + c] =
      src[(size_t)blockIdx.y * C * N + e];
}

__global__ void k_gather(const float* __restrict__ src, const int* __restrict__ idx,
                         float* __restrict__ dst, int Nin, int Nout, int D) {
  int e = blockIdx.x * blockDim.x + threadIdx.x;
  if (e >= Nout * D) return;
  int row = e / D, d = e % D;
  int s = idx[(size_t)blockIdx.y * Nout + row];
  dst[(size_t)blockIdx.y * Nout * D + e] =
      src[(size_t)blockIdx.y * Nin * D + (size_t)s * D + d];
}

__global__ void k_interp4_concat(const float* __restrict__ x, const float* __restrict__ skip,
                                 float* __restrict__ out, int n, int Dx, int Ds) {
  const int Do = Dx + Ds;
  int e = blockIdx.x * blockDim.x + threadIdx.x;
  if (e >= 4 * n * Do) return;
  int j = e / Do, d = e % Do;
  float v;
  if (d < Dx) {
    float pos = fminf(fmaxf((j + 0.5f) * 0.25f - 0.5f, 0.f), (float)(n - 1));
    int i0 = (int)floorf(pos);
    int i1 = min(i0 + 1, n - 1);
    float w = pos - (float)i0;
    v = x[((size_t)blockIdx.y * n + i0) * Dx + d] * (1.f - w) +
        x[((size_t)blockIdx.y * n + i1) * Dx + d] * w;
  } else {
    v = skip[((size_t)blockIdx.y * 4 * n + j) * Ds + (d - Dx)];
  }
  out[((size_t)blockIdx.y * 4 * n + j) * Do + d] = v;
}

// ---------------------------------------------------------------------------
// Host helpers
// ---------------------------------------------------------------------------
struct Lin { const float* W; const float* b; };
struct TLW { Lin f, k, q, v; };

template <int K, int Nc, bool OH>
static inline void launch_gemm(const float* A, const Lin& L, const float* resid,
                               void* C, int M, int storeT, int Bb, hipStream_t s) {
  dim3 g(M / 16, Nc / 16, Bb);
  k_gemm_nt<K, Nc, OH><<<g, 32, 0, s>>>(A, L.W, L.b, resid, C, M, storeT);
}

template <int D>
static inline void run_attn(const half_t* q, const half_t* k, const half_t* v,
                            float* o, int Np, int Bb, hipStream_t s) {
  dim3 g(Np / 64, Bb);
  size_t sh = (size_t)(256 * D + 4096);  // 2 bufs * (K+V) * 32*D f16 + P staging
  k_attn<D><<<g, 128, sh, s>>>(q, k, v, o, Np);
}

template <int D>
static inline void run_tlayer(const float* x, const TLW& t, float* out, int Np, int Bb,
                              half_t* tq, half_t* tk, half_t* tv, float* to,
                              hipStream_t s) {
  launch_gemm<D, D, true>(x, t.q, nullptr, tq, Np, 0, Bb, s);
  launch_gemm<D, D, true>(x, t.k, nullptr, tk, Np, 0, Bb, s);
  launch_gemm<D, D, true>(x, t.v, nullptr, tv, Np, 0, Bb, s);
  run_attn<D>(tq, tk, tv, to, Np, Bb, s);
  launch_gemm<D, D, false>(to, t.f, /*resid=*/x, out, Np, 0, Bb, s);
}

// ---------------------------------------------------------------------------
// Entry point
// ---------------------------------------------------------------------------
extern "C" void kernel_launch(void* const* d_in, const int* in_sizes, int n_in,
                              void* d_out, int out_size, void* d_ws, size_t ws_size,
                              hipStream_t stream) {
  (void)in_sizes; (void)n_in; (void)out_size; (void)ws_size;
  const int Bb = 4, N0 = 16384, N1 = 4096, N2 = 1024, CIN = 32;

  const float* in_points = (const float*)d_in[0];
  const float* in_features = (const float*)d_in[1];
  auto P = [&](int i) { return (const float*)d_in[i]; };
  // JAX pytree leaf order: top-level insertion (points, features, params);
  // nested dict keys sorted: top {conv0,ds0,ds1,enc0,enc1,final,up0,up1},
  // lin {W,b}, tlayer {f,k,q,v}, up {c1,c2,t1,t2}.
  Lin conv0{P(2), P(3)}, ds0{P(4), P(5)}, ds1{P(6), P(7)};
  TLW enc0{{P(8), P(9)}, {P(10), P(11)}, {P(12), P(13)}, {P(14), P(15)}};
  TLW enc1{{P(16), P(17)}, {P(18), P(19)}, {P(20), P(21)}, {P(22), P(23)}};
  Lin finalL{P(24), P(25)};
  Lin up0c1{P(26), P(27)}, up0c2{P(28), P(29)};
  TLW up0t1{{P(30), P(31)}, {P(32), P(33)}, {P(34), P(35)}, {P(36), P(37)}};
  TLW up0t2{{P(38), P(39)}, {P(40), P(41)}, {P(42), P(43)}, {P(44), P(45)}};
  Lin up1c1{P(46), P(47)}, up1c2{P(48), P(49)};
  TLW up1t1{{P(50), P(51)}, {P(52), P(53)}, {P(54), P(55)}, {P(56), P(57)}};
  TLW up1t2{{P(58), P(59)}, {P(60), P(61)}, {P(62), P(63)}, {P(64), P(65)}};

  char* wsp = (char*)d_ws;
  size_t off = 0;
  auto allocF = [&](size_t n) { float* p = (float*)(wsp + off); off += ((n * 4 + 255) / 256) * 256; return p; };
  auto allocI = [&](size_t n) { int* p = (int*)(wsp + off); off += ((n * 4 + 255) / 256) * 256; return p; };

  float* ptsT0 = allocF((size_t)Bb * N0 * 3);
  float* fT    = allocF((size_t)Bb * N0 * CIN);
  float* x0    = allocF((size_t)Bb * N0 * 128);   // skip0
  float* ptsT1 = allocF((size_t)Bb * N1 * 3);
  float* s1    = allocF((size_t)Bb * N1 * 128);   // skip1
  float* cat1  = allocF((size_t)Bb * N1 * 384);
  float* cat0  = allocF((size_t)Bb * N0 * 256);
  half_t* tq = (half_t*)allocF((size_t)Bb * N1 * 64);  // f16 [B,Np,D]
  half_t* tk = (half_t*)allocF((size_t)Bb * N1 * 64);
  half_t* tv = (half_t*)allocF((size_t)Bb * N1 * 64);
  float* to = allocF((size_t)Bb * N1 * 128);
  float* pa = allocF((size_t)Bb * N1 * 128);      // ping
  float* pb = allocF((size_t)Bb * N1 * 128);      // pong
  int* idx0 = allocI((size_t)Bb * N1);
  int* idx1 = allocI((size_t)Bb * N2);

  hipFuncSetAttribute(reinterpret_cast<const void*>(&k_fps),
                      hipFuncAttributeMaxDynamicSharedMemorySize, 4 * N0 * 4);
  hipFuncSetAttribute(reinterpret_cast<const void*>(&k_attn<128>),
                      hipFuncAttributeMaxDynamicSharedMemorySize, 256 * 128 + 4096);
  hipFuncSetAttribute(reinterpret_cast<const void*>(&k_attn<256>),
                      hipFuncAttributeMaxDynamicSharedMemorySize, 256 * 256 + 4096);

  auto transpose = [&](const float* s, float* d, int C, int N) {
    dim3 g((C * N + 255) / 256, Bb);
    k_transpose_cn<<<g, 256, 0, stream>>>(s, d, C, N);
  };
  auto gather = [&](const float* s, const int* ix, float* d, int Nin, int Nout, int D) {
    dim3 g((Nout * D + 255) / 256, Bb);
    k_gather<<<g, 256, 0, stream>>>(s, ix, d, Nin, Nout, D);
  };
  auto fps = [&](const float* pts, int Np, int npoint, int* ix) {
    k_fps<<<dim3(Bb), 1024, (size_t)4 * Np * sizeof(float), stream>>>(pts, Np, npoint, ix);
  };
  auto interp_cat = [&](const float* x, const float* skip, float* out, int n, int Dx, int Ds) {
    dim3 g((4 * n * (Dx + Ds) + 255) / 256, Bb);
    k_interp4_concat<<<g, 256, 0, stream>>>(x, skip, out, n, Dx, Ds);
  };

  // ---- pipeline ----
  transpose(in_points, ptsT0, 3, N0);
  transpose(in_features, fT, CIN, N0);
  launch_gemm<32, 128, false>(fT, conv0, nullptr, x0, N0, 0, Bb, stream);   // conv0

  fps(ptsT0, N0, N1, idx0);
  gather(ptsT0, idx0, ptsT1, N0, N1, 3);
  gather(x0, idx0, pa, N0, N1, 128);
  launch_gemm<128, 128, false>(pa, ds0, nullptr, pb, N1, 0, Bb, stream);    // ds0
  run_tlayer<128>(pb, enc0, s1, N1, Bb, tq, tk, tv, to, stream);            // enc0

  fps(ptsT1, N1, N2, idx1);
  gather(s1, idx1, pa, N1, N2, 128);
  launch_gemm<128, 256, false>(pa, ds1, nullptr, pb, N2, 0, Bb, stream);    // ds1
  run_tlayer<256>(pb, enc1, pa, N2, Bb, tq, tk, tv, to, stream);            // enc1

  launch_gemm<256, 256, false>(pa, up1c1, nullptr, pb, N2, 0, Bb, stream);  // up1.c1
  run_tlayer<256>(pb, up1t1, pa, N2, Bb, tq, tk, tv, to, stream);           // up1.t1
  launch_gemm<256, 256, false>(pa, up1c2, nullptr, pb, N2, 0, Bb, stream);  // up1.c2
  run_tlayer<256>(pb, up1t2, pa, N2, Bb, tq, tk, tv, to, stream);           // up1.t2
  interp_cat(pa, s1, cat1, N2, 256, 128);                                   // -> [B,N1,384]

  launch_gemm<384, 128, false>(cat1, up0c1, nullptr, pa, N1, 0, Bb, stream);// up0.c1
  run_tlayer<128>(pa, up0t1, pb, N1, Bb, tq, tk, tv, to, stream);           // up0.t1
  launch_gemm<128, 128, false>(pb, up0c2, nullptr, pa, N1, 0, Bb, stream);  // up0.c2
  run_tlayer<128>(pa, up0t2, pb, N1, Bb, tq, tk, tv, to, stream);           // up0.t2
  interp_cat(pb, x0, cat0, N1, 128, 128);                                   // -> [B,N0,256]

  launch_gemm<256, 64, false>(cat0, finalL, nullptr, (float*)d_out, N0, 1, Bb, stream);
}
